// Attention_34076270526817
// MI455X (gfx1250) — compile-verified, matrix operand order checked by reading, not run on hardware
//
#include <hip/hip_runtime.h>
#include <hip/hip_bf16.h>
#include <math.h>

typedef __bf16 v16bf __attribute__((ext_vector_type(16)));
typedef __bf16 v8bf  __attribute__((ext_vector_type(8)));
typedef float  v8f   __attribute__((ext_vector_type(8)));

#define BB_  2
#define LL_  4
#define NN_  1024
#define DIM_ 512
#define HEADS_ 8
#define DH_  64
#define INNER_ 512
#define NS_  8
#define ROWS_ (BB_*LL_*NN_)   /* 8192 */
#define QKVN_ (3*INNER_)      /* 1536 */

// ---------------- wave reductions (wave32) ----------------
static __device__ __forceinline__ float wave_max_f(float v) {
#pragma unroll
  for (int m = 16; m > 0; m >>= 1) v = fmaxf(v, __shfl_xor(v, m));
  return v;
}
static __device__ __forceinline__ float wave_sum_f(float v) {
#pragma unroll
  for (int m = 16; m > 0; m >>= 1) v += __shfl_xor(v, m);
  return v;
}

// ---------------- WMMA bf16 16x16x32 fragments ----------------
// Per-lane fragment = 16 bf16 = two contiguous 16-byte loads.
// lanes 0-15: K = {0..7, 16..23}; lanes 16-31: K = {8..15, 24..31}
static __device__ __forceinline__ v16bf load_frag16(const __bf16* p) {
  v8bf lo = *reinterpret_cast<const v8bf*>(p);
  v8bf hi = *reinterpret_cast<const v8bf*>(p + 16);
  v16bf r;
#pragma unroll
  for (int e = 0; e < 8; ++e) { r[e] = lo[e]; r[e + 8] = hi[e]; }
  return r;
}

static __device__ __forceinline__ v8f wmma_bf16(v16bf a, v16bf b, v8f c) {
  return __builtin_amdgcn_wmma_f32_16x16x32_bf16(false, a, false, b, (short)0, c,
                                                 false, false);
}

// C strip (16 x 64, f32) = A(16xK) * B^T(64xK); 4 accumulators, A reused 4x.
// Explicit software pipeline: the NEXT k-step's 10 b128 loads are issued before
// the CURRENT step's 4 WMMAs, with two full fragment buffer sets, so the WMMA
// group of step i only waits on step-i loads while step-(i+1) loads stream.
template <int K>
static __device__ __forceinline__ void gemm_strip_bf16(const __bf16* __restrict__ A,
                                                       const __bf16* __restrict__ Bt,
                                                       int tileM, int tileN,
                                                       int lane, v8f c[4]) {
  const int mk    = lane & 15;
  const int khalf = lane >> 4;
  const __bf16* ap = A  + (size_t)(tileM + mk) * K + khalf * 8;
  const __bf16* bp = Bt + (size_t)(tileN + mk) * K + khalf * 8;
#pragma unroll
  for (int t = 0; t < 4; ++t) c[t] = (v8f){};

  v16bf a  = load_frag16(ap);
  v16bf b0 = load_frag16(bp + 0 * 16 * K);
  v16bf b1 = load_frag16(bp + 1 * 16 * K);
  v16bf b2 = load_frag16(bp + 2 * 16 * K);
  v16bf b3 = load_frag16(bp + 3 * 16 * K);

#pragma unroll
  for (int k0 = 0; k0 < K; k0 += 32) {
    const int kn = k0 + 32;
    v16bf an, bn0, bn1, bn2, bn3;
    if (kn < K) {                       // prefetch next step (compile-time pred)
      an  = load_frag16(ap + kn);
      bn0 = load_frag16(bp + 0 * 16 * K + kn);
      bn1 = load_frag16(bp + 1 * 16 * K + kn);
      bn2 = load_frag16(bp + 2 * 16 * K + kn);
      bn3 = load_frag16(bp + 3 * 16 * K + kn);
    }
    c[0] = wmma_bf16(a, b0, c[0]);
    c[1] = wmma_bf16(a, b1, c[1]);
    c[2] = wmma_bf16(a, b2, c[2]);
    c[3] = wmma_bf16(a, b3, c[3]);
    if (kn < K) { a = an; b0 = bn0; b1 = bn1; b2 = bn2; b3 = bn3; }
  }
}

// ---------------- 1) LayerNorm(512) -> bf16 ----------------
__global__ void k_layernorm(const float* __restrict__ f, const float* __restrict__ gamma,
                            const float* __restrict__ beta, __bf16* __restrict__ out) {
  __shared__ float s_sum[256];
  __shared__ float s_sq[256];
  const int p   = blockIdx.x;
  const int tid = threadIdx.x;
  const float* fp = f + (size_t)p * DIM_;
  float x0 = fp[tid], x1 = fp[tid + 256];
  s_sum[tid] = x0 + x1;
  s_sq[tid]  = x0 * x0 + x1 * x1;
  __syncthreads();
#pragma unroll
  for (int s = 128; s > 0; s >>= 1) {
    if (tid < s) { s_sum[tid] += s_sum[tid + s]; s_sq[tid] += s_sq[tid + s]; }
    __syncthreads();
  }
  const float mu  = s_sum[0] * (1.0f / DIM_);
  const float var = s_sq[0] * (1.0f / DIM_) - mu * mu;
  const float rs  = rsqrtf(var + 1e-5f);
  __bf16* op = out + (size_t)p * DIM_;
  op[tid]       = (__bf16)((x0 - mu) * rs * gamma[tid]       + beta[tid]);
  op[tid + 256] = (__bf16)((x1 - mu) * rs * gamma[tid + 256] + beta[tid + 256]);
}

// ---------------- 2) weight convert + transpose (K-major) ----------------
__global__ void k_prep_w(const float* __restrict__ w, __bf16* __restrict__ wT, int ncols) {
  const int t  = blockIdx.x * blockDim.x + threadIdx.x;  // ncols*512 threads
  const int kc = t & (DIM_ - 1);
  const int nc = t >> 9;
  wT[(size_t)nc * DIM_ + kc] = (__bf16)w[(size_t)kc * ncols + nc];
}

// ---------------- 3) ball query: first 8 in-radius (CUDA pointnet2 semantics) ----
__global__ void k_ballquery(const float* __restrict__ xyzs, int* __restrict__ idxb) {
  const int t  = blockIdx.x * blockDim.x + threadIdx.x;  // L*B*L*N = 32768
  const int qn = t & (NN_ - 1);
  const int ll = (t >> 10) & (LL_ - 1);
  const int bb = (t >> 12) & (BB_ - 1);
  const int fi = t >> 13;
  const float* qx = xyzs + (size_t)((bb * LL_ + fi) * NN_ + qn) * 3;
  const float qx0 = qx[0], qx1 = qx[1], qx2 = qx[2];
  const float* ref = xyzs + (size_t)((bb * LL_ + ll) * NN_) * 3;
  int found[NS_];
  int cnt = 0;
  for (int r = 0; r < NN_ && cnt < NS_; ++r) {
    const float dx = ref[r * 3 + 0] - qx0;
    const float dy = ref[r * 3 + 1] - qx1;
    const float dz = ref[r * 3 + 2] - qx2;
    if (dx * dx + dy * dy + dz * dz < 0.04f) found[cnt++] = r;
  }
  const int first = (cnt > 0) ? found[0] : 0;
  int* op = idxb + (size_t)t * NS_;
#pragma unroll
  for (int s = 0; s < NS_; ++s) op[s] = (s < cnt) ? found[s] : first;
}

// ---------------- 4) qkv GEMM: (8192x512) @ (512x1536), WMMA bf16 ----------------
// One wave = 16x64 output strip (4 accumulators).
__global__ void k_gemm_qkv(const __bf16* __restrict__ A, const __bf16* __restrict__ Bt,
                           float* __restrict__ C) {
  const int lane = threadIdx.x & 31;
  const int w    = (blockIdx.x * blockDim.x + threadIdx.x) >> 5;
  const int tn64 = QKVN_ / 64;                 // 24
  const int tileN = (w % tn64) * 64;
  const int tileM = (w / tn64) * 16;
  v8f c[4];
  gemm_strip_bf16<DIM_>(A, Bt, tileM, tileN, lane, c);
  const int rbase = tileM + ((lane >> 4) << 3);
#pragma unroll
  for (int t = 0; t < 4; ++t) {
    const int col = tileN + t * 16 + (lane & 15);
#pragma unroll
    for (int r = 0; r < 8; ++r) C[(size_t)(rbase + r) * QKVN_ + col] = c[t][r];
  }
}

// ---------------- 5) attention: one wave per (frame,batch,point,head) ----------
__global__ void k_attention(const float* __restrict__ qkv, const float* __restrict__ xyzs,
                            const int* __restrict__ idxb, const float* __restrict__ wsp,
                            __bf16* __restrict__ fr) {
  const int lane = threadIdx.x & 31;
  const int wid  = (blockIdx.x * blockDim.x + threadIdx.x) >> 5;
  const int h  = wid & (HEADS_ - 1);
  const int qn = (wid >> 3) & (NN_ - 1);
  const int bb = (wid >> 13) & (BB_ - 1);
  const int fi = wid >> 14;

  // lane j = key j: frame fl = j/8, sample s = j%8
  const int fl = lane >> 3, s = lane & 7;
  const int nb = idxb[((((size_t)fi * BB_ + bb) * LL_ + fl) * NN_ + qn) * NS_ + s];
  const int rowq = (bb * LL_ + fi) * NN_ + qn;
  const int rowk = (bb * LL_ + fl) * NN_ + nb;

  const float* qp = qkv + (size_t)rowq * QKVN_ + h * DH_;
  const float* kp = qkv + (size_t)rowk * QKVN_ + INNER_ + h * DH_;
  float logit = 0.f;
#pragma unroll 8
  for (int d = 0; d < DH_; ++d) logit += qp[d] * kp[d];
  logit *= 0.125f;  // dh^-0.5

  const float m    = wave_max_f(logit);
  const float e    = __expf(logit - m);
  const float ssum = wave_sum_f(e);
  const float attn = e / ssum;

  // dis_attn = max_j attn_j * disp_j  (per coord), wave-uniform after reduce
  const float* gx = xyzs + (size_t)rowk * 3;
  const float* qx = xyzs + (size_t)rowq * 3;
  float dmax[3];
#pragma unroll
  for (int c = 0; c < 3; ++c) dmax[c] = wave_max_f(attn * (gx[c] - qx[c]));

  // attnout: lane handles dims d0=lane, d1=lane+32
  float acc0 = 0.f, acc1 = 0.f;
  for (int j = 0; j < 32; ++j) {
    const float aj = __shfl(attn, j);
    const int   rj = __shfl(rowk, j);
    const float* vp = qkv + (size_t)rj * QKVN_ + 2 * INNER_ + h * DH_;
    acc0 += aj * vp[lane];
    acc1 += aj * vp[lane + 32];
  }
  // + (dis_attn @ w_spatial)  with w_spatial (3,64)
  acc0 += dmax[0] * wsp[lane]      + dmax[1] * wsp[64 + lane] + dmax[2] * wsp[128 + lane];
  acc1 += dmax[0] * wsp[lane + 32] + dmax[1] * wsp[96 + lane] + dmax[2] * wsp[160 + lane];

  __bf16* op = fr + (size_t)rowq * INNER_ + h * DH_;
  op[lane]      = (__bf16)acc0;
  op[lane + 32] = (__bf16)acc1;
}

// ---------------- 6) out GEMM + bias + exact GELU + residual ----------------
__global__ void k_gemm_out(const __bf16* __restrict__ A, const __bf16* __restrict__ Bt,
                           const float* __restrict__ bias, const float* __restrict__ feat,
                           float* __restrict__ out) {
  const int lane = threadIdx.x & 31;
  const int w    = (blockIdx.x * blockDim.x + threadIdx.x) >> 5;
  const int tn64 = DIM_ / 64;                  // 8
  const int tileN = (w & (tn64 - 1)) * 64;
  const int tileM = (w / tn64) * 16;
  v8f c[4];
  gemm_strip_bf16<DIM_>(A, Bt, tileM, tileN, lane, c);
  const int rbase = tileM + ((lane >> 4) << 3);
#pragma unroll
  for (int t = 0; t < 4; ++t) {
    const int col = tileN + t * 16 + (lane & 15);
    const float bv = bias[col];
#pragma unroll
    for (int r = 0; r < 8; ++r) {
      const int row = rbase + r;
      const float x = c[t][r] + bv;
      const float g = 0.5f * x * (1.0f + erff(x * 0.70710678118654752f));
      out[(size_t)row * DIM_ + col] = g + feat[(size_t)row * DIM_ + col];
    }
  }
}

// ---------------- launch ----------------
extern "C" void kernel_launch(void* const* d_in, const int* in_sizes, int n_in,
                              void* d_out, int out_size, void* d_ws, size_t ws_size,
                              hipStream_t stream) {
  (void)in_sizes; (void)n_in; (void)out_size; (void)ws_size;
  const float* xyzs    = (const float*)d_in[0];
  const float* feature = (const float*)d_in[1];
  const float* gamma   = (const float*)d_in[2];
  const float* beta    = (const float*)d_in[3];
  const float* w_qkv   = (const float*)d_in[4];
  const float* w_sp    = (const float*)d_in[5];
  const float* w_out   = (const float*)d_in[6];
  const float* b_out   = (const float*)d_in[7];
  float* out = (float*)d_out;

  char* ws = (char*)d_ws;
  size_t off = 0;
  __bf16* normf = (__bf16*)(ws + off); off += (size_t)ROWS_ * DIM_ * 2;        // 8 MB
  __bf16* wqkvT = (__bf16*)(ws + off); off += (size_t)QKVN_ * DIM_ * 2;        // 1.5 MB
  __bf16* woutT = (__bf16*)(ws + off); off += (size_t)DIM_ * DIM_ * 2;         // 0.5 MB
  float*  qkv   = (float*)(ws + off);  off += (size_t)ROWS_ * QKVN_ * 4;       // 48 MB
  int*    idxb  = (int*)(ws + off);    off += (size_t)LL_ * BB_ * LL_ * NN_ * NS_ * 4; // 1 MB
  __bf16* fr    = (__bf16*)(ws + off); off += (size_t)ROWS_ * INNER_ * 2;      // 8 MB

  k_layernorm<<<ROWS_, 256, 0, stream>>>(feature, gamma, beta, normf);
  k_prep_w<<<(QKVN_ * DIM_) / 256, 256, 0, stream>>>(w_qkv, wqkvT, QKVN_);
  k_prep_w<<<(DIM_ * DIM_) / 256, 256, 0, stream>>>(w_out, woutT, DIM_);
  k_ballquery<<<(LL_ * BB_ * LL_ * NN_) / 256, 256, 0, stream>>>(xyzs, idxb);
  // (8192/16) x (1536/64) = 12288 strip-waves, 8 waves/block
  k_gemm_qkv<<<(ROWS_ / 16) * (QKVN_ / 64) / 8, 256, 0, stream>>>(normf, wqkvT, qkv);
  // 65536 (point,head) waves, 8 waves/block
  k_attention<<<(LL_ * BB_ * NN_ * HEADS_) / 8, 256, 0, stream>>>(qkv, xyzs, idxb, w_sp, fr);
  // (8192/16) x (512/64) = 4096 strip-waves
  k_gemm_out<<<(ROWS_ / 16) * (DIM_ / 64) / 8, 256, 0, stream>>>(fr, woutT, b_out, feature, out);
}
